// DummyRNN_23029614641152
// MI455X (gfx1250) — compile-verified, matrix operand order checked by reading, not run on hardware
//
#include <hip/hip_runtime.h>

// ---------------------------------------------------------------------------
// Elman RNN for MI455X (gfx1250): h_t = tanh(x_t Ww^T + h_{t-1} Uw^T + bias)
// Phase 0: convert X/Ww/Uw f32 -> bf16 (streaming), zero h0
// Phase 1: big parallel GEMM Wx = X*Ww^T + bias  (bf16 WMMA, f32 accumulate,
//          async-to-LDS tile staging)
// Phase 2: 512 sequential step kernels; Uw bf16 resident in 192MB L2;
//          double-buffered GLOBAL_LOAD_ASYNC_TO_LDS staging overlapped with
//          v_wmma compute; fused tanh epilogue.
// ---------------------------------------------------------------------------

typedef __attribute__((ext_vector_type(16))) __bf16 v16bf;
typedef __attribute__((ext_vector_type(8)))  float  v8f;

#define TT     512
#define BB     64
#define DIN    1024
#define DOUT   2048

#define LDSTR  72     // 64 + 8 bf16 pad -> 144B row stride (16B aligned)

union FragCast { v16bf v; uint4 q[2]; };

__device__ __forceinline__ v8f wmma_bf16(v16bf a, v16bf b, v8f c) {
  // (neg_a, A, neg_b, B, c_mod, C, reuse_a, reuse_b)
  return __builtin_amdgcn_wmma_f32_16x16x32_bf16(
      false, a, false, b, (short)0, c, false, false);
}

// Async DMA: 16B per lane, global -> LDS, tracked by ASYNCcnt.
__device__ __forceinline__ void async_b128(__bf16* lds_p, const __bf16* gp) {
  const uint32_t l = (uint32_t)(uintptr_t)lds_p;   // low 32b = LDS offset
  const uint64_t g = (uint64_t)(uintptr_t)gp;
  asm volatile("global_load_async_to_lds_b128 %0, %1, off"
               :: "v"(l), "v"(g) : "memory");
}

__device__ __forceinline__ void wait_async0() {
  asm volatile("s_wait_asynccnt 0x0" ::: "memory");
}

// Load one 16x32 bf16 fragment (A: lane=row / B: lane=col, VGPRs stripe K).
// Per ISA 7.12.2: lanes 0-15 hold K[0..7] (v0..3) and K[16..23] (v4..7),
// lanes 16-31 hold K[8..15] and K[24..31] for the same row/col index.
__device__ __forceinline__ v16bf load_frag_lds(const __bf16* smem, int base,
                                               int kBase) {
  const int lane = threadIdx.x & 31;
  const int hl   = lane >> 4;
  const int r    = base + (lane & 15);
  const __bf16* p = smem + r * LDSTR + kBase + 8 * hl;
  FragCast f;
  f.q[0] = *(const uint4*)(p);
  f.q[1] = *(const uint4*)(p + 16);
  return f.v;
}

// ---------------------------------------------------------------------------
// Phase 0a: f32 -> bf16 streaming convert (8 elements/thread)
// ---------------------------------------------------------------------------
__global__ __launch_bounds__(256) void cvt_f32_bf16_kernel(
    const float* __restrict__ src, __bf16* __restrict__ dst) {
  const size_t i = ((size_t)blockIdx.x * blockDim.x + threadIdx.x) * 8;
  const float4 a = *(const float4*)(src + i);
  const float4 b = *(const float4*)(src + i + 4);
  __bf16 o[8] = { (__bf16)a.x, (__bf16)a.y, (__bf16)a.z, (__bf16)a.w,
                  (__bf16)b.x, (__bf16)b.y, (__bf16)b.z, (__bf16)b.w };
  *(uint4*)(dst + i) = *(const uint4*)o;
}

// ---------------------------------------------------------------------------
// Phase 0b: zero initial hidden state (bf16 [BB,DOUT])
// ---------------------------------------------------------------------------
__global__ __launch_bounds__(256) void zero_h_kernel(uint32_t* __restrict__ p) {
  p[(size_t)blockIdx.x * blockDim.x + threadIdx.x] = 0u;
}

// ---------------------------------------------------------------------------
// Phase 1: Wx[m,n] = sum_k Xb[m,k]*Wwb[n,k] + (Wb[n]+Ub[n]+b[n])
//   M = 32768, K = 1024, N = 2048.  WG tile 128x128, K-tile 64,
//   8 waves in 2x4, wave tile 64x32.  Tiles staged via async-to-LDS.
// ---------------------------------------------------------------------------
__global__ __launch_bounds__(256) void gemm_wx_kernel(
    const __bf16* __restrict__ Xb, const __bf16* __restrict__ Wwb,
    const float* __restrict__ Wb, const float* __restrict__ Ub,
    const float* __restrict__ bv, float* __restrict__ Wx) {
  __shared__ alignas(16) __bf16 As[128 * LDSTR];
  __shared__ alignas(16) __bf16 Bs[128 * LDSTR];

  const int m0   = blockIdx.x * 128;
  const int n0   = blockIdx.y * 128;
  const int tid  = threadIdx.x;
  const int w    = tid >> 5;
  const int lane = tid & 31;
  const int wr   = w >> 2;   // 0..1  (64 rows each)
  const int wc   = w & 3;    // 0..3  (32 cols each)

  v8f zero = {};
  v8f acc[4][2];
#pragma unroll
  for (int i = 0; i < 4; ++i)
#pragma unroll
    for (int j = 0; j < 2; ++j) acc[i][j] = zero;

  for (int k0 = 0; k0 < DIN; k0 += 64) {
    // async stage: A = Xb[m0..+127][k0..+63], B = Wwb[n0..+127][k0..+63]
    // 128 rows x 8 chunks(16B) = 1024 chunks each -> 4 per thread per matrix
#pragma unroll
    for (int i = 0; i < 4; ++i) {
      const int c   = tid + 256 * i;     // 0..1023
      const int r   = c >> 3;            // 0..127
      const int col = (c & 7) * 8;       // bf16 column
      async_b128(As + r * LDSTR + col, Xb  + (size_t)(m0 + r) * DIN + k0 + col);
      async_b128(Bs + r * LDSTR + col, Wwb + (size_t)(n0 + r) * DIN + k0 + col);
    }
    wait_async0();
    __syncthreads();
#pragma unroll
    for (int kk = 0; kk < 64; kk += 32) {
      v16bf af[4], bf[2];
#pragma unroll
      for (int i = 0; i < 4; ++i) af[i] = load_frag_lds(As, wr * 64 + i * 16, kk);
#pragma unroll
      for (int j = 0; j < 2; ++j) bf[j] = load_frag_lds(Bs, wc * 32 + j * 16, kk);
#pragma unroll
      for (int i = 0; i < 4; ++i)
#pragma unroll
        for (int j = 0; j < 2; ++j)
          acc[i][j] = wmma_bf16(af[i], bf[j], acc[i][j]);
    }
    __syncthreads();
  }

  // epilogue: C/D layout (7.12.2): lanes 0-15 -> N=lane, rows 0..7 in v0..7;
  // lanes 16-31 -> rows 8..15
  const int hl   = lane >> 4;
  const int nloc = lane & 15;
#pragma unroll
  for (int j = 0; j < 2; ++j) {
    const int n = n0 + wc * 32 + j * 16 + nloc;
    const float bias = Wb[n] + Ub[n] + bv[n];
#pragma unroll
    for (int i = 0; i < 4; ++i) {
      const int mb = m0 + wr * 64 + i * 16 + 8 * hl;
      float* o = Wx + (size_t)mb * DOUT + n;
#pragma unroll
      for (int v = 0; v < 8; ++v) o[(size_t)v * DOUT] = acc[i][j][v] + bias;
    }
  }
}

// ---------------------------------------------------------------------------
// Phase 2: one timestep.  acc = Hprev * Uwb^T  (64x2048x2048 bf16 WMMA),
//   h = tanh(acc + Wx_t) -> f32 out, bf16 Hcur.
//   Grid: 16 WGs (N tiles of 128). 8 waves; wave tile 64(M) x 16(N).
//   Double-buffered async-to-LDS staging overlapped with WMMA compute.
// ---------------------------------------------------------------------------
__global__ __launch_bounds__(256) void rnn_step_kernel(
    const float* __restrict__ WxT,     // [64, DOUT] preactivation (bias inside)
    const __bf16* __restrict__ Hprev,  // [64, DOUT]
    const __bf16* __restrict__ Uwb,    // [DOUT, DOUT] bf16 (L2 resident)
    float* __restrict__ OutT,          // [64, DOUT]
    __bf16* __restrict__ Hcur) {       // [64, DOUT]
  __shared__ alignas(16) __bf16 As[2][64 * LDSTR];    // 2 x 9KB
  __shared__ alignas(16) __bf16 Bs[2][128 * LDSTR];   // 2 x 18KB

  const int n0   = blockIdx.x * 128;
  const int tid  = threadIdx.x;
  const int w    = tid >> 5;   // 0..7 -> N sub-tile
  const int lane = tid & 31;

  v8f zero = {};
  v8f acc[4];
#pragma unroll
  for (int i = 0; i < 4; ++i) acc[i] = zero;

  // stage tile (A: 64x64 = 512 16B-chunks -> 2/thread;
  //             B: 128x64 = 1024 chunks -> 4/thread)
  auto stage = [&](int buf, int k0) {
#pragma unroll
    for (int i = 0; i < 2; ++i) {
      const int c   = tid + 256 * i;   // 0..511
      const int r   = c >> 3;          // 0..63
      const int col = (c & 7) * 8;
      async_b128(&As[buf][r * LDSTR + col],
                 Hprev + (size_t)r * DOUT + k0 + col);
    }
#pragma unroll
    for (int i = 0; i < 4; ++i) {
      const int c   = tid + 256 * i;   // 0..1023
      const int r   = c >> 3;          // 0..127
      const int col = (c & 7) * 8;
      async_b128(&Bs[buf][r * LDSTR + col],
                 Uwb + (size_t)(n0 + r) * DOUT + k0 + col);
    }
  };

  stage(0, 0);
  int buf = 0;
  for (int k0 = 0; k0 < DOUT; k0 += 64) {
    wait_async0();        // tile[buf] landed (this wave's DMAs)
    __syncthreads();      // ... and every wave's DMAs
    if (k0 + 64 < DOUT) stage(buf ^ 1, k0 + 64);   // overlap next tile DMA
#pragma unroll
    for (int kk = 0; kk < 64; kk += 32) {
      const v16bf bf = load_frag_lds(Bs[buf], w * 16, kk);
#pragma unroll
      for (int i = 0; i < 4; ++i) {
        const v16bf af = load_frag_lds(As[buf], i * 16, kk);
        acc[i] = wmma_bf16(af, bf, acc[i]);
      }
    }
    __syncthreads();      // everyone done reading tile[buf]
    buf ^= 1;
  }

  const int hl = lane >> 4;
  const int n  = n0 + w * 16 + (lane & 15);
#pragma unroll
  for (int i = 0; i < 4; ++i) {
    const int mb = i * 16 + 8 * hl;
#pragma unroll
    for (int v = 0; v < 8; ++v) {
      const size_t idx = (size_t)(mb + v) * DOUT + n;
      const float h = tanhf(acc[i][v] + WxT[idx]);
      OutT[idx] = h;
      Hcur[idx] = (__bf16)h;
    }
  }
}

// ---------------------------------------------------------------------------
extern "C" void kernel_launch(void* const* d_in, const int* in_sizes, int n_in,
                              void* d_out, int out_size, void* d_ws,
                              size_t ws_size, hipStream_t stream) {
  (void)in_sizes; (void)n_in; (void)out_size; (void)ws_size;
  const float* X  = (const float*)d_in[0];   // [T,B,DIN]
  const float* Ww = (const float*)d_in[1];   // [DOUT,DIN]
  const float* Wb = (const float*)d_in[2];   // [DOUT]
  const float* Uw = (const float*)d_in[3];   // [DOUT,DOUT]
  const float* Ub = (const float*)d_in[4];   // [DOUT]
  const float* bv = (const float*)d_in[5];   // [DOUT]
  float* out = (float*)d_out;                // [T*B, DOUT]

  char* ws = (char*)d_ws;
  size_t off = 0;
  float*  Wx  = (float*)(ws + off);  off += (size_t)TT * BB * DOUT * 4; // 256MB
  __bf16* Uwb = (__bf16*)(ws + off); off += (size_t)DOUT * DOUT * 2;    // 8MB
  __bf16* Xb  = (__bf16*)(ws + off); off += (size_t)TT * BB * DIN * 2;  // 64MB
  __bf16* Wwb = (__bf16*)(ws + off); off += (size_t)DOUT * DIN * 2;     // 4MB
  __bf16* H0  = (__bf16*)(ws + off); off += (size_t)BB * DOUT * 2;
  __bf16* H1  = (__bf16*)(ws + off);

  // Phase 0: bf16 conversions (one streaming pass each) + h0 = 0
  cvt_f32_bf16_kernel<<<(DOUT * DOUT) / (8 * 256), 256, 0, stream>>>(Uw, Uwb);
  cvt_f32_bf16_kernel<<<((size_t)TT * BB * DIN) / (8 * 256), 256, 0, stream>>>(X, Xb);
  cvt_f32_bf16_kernel<<<(DOUT * DIN) / (8 * 256), 256, 0, stream>>>(Ww, Wwb);
  zero_h_kernel<<<(BB * DOUT / 2) / 256, 256, 0, stream>>>((uint32_t*)H0);

  // Phase 1: Wx = X*Ww^T + (Wb+Ub+b)   [32768 x 2048]
  dim3 grid((TT * BB) / 128, DOUT / 128);
  gemm_wx_kernel<<<grid, 256, 0, stream>>>(Xb, Wwb, Wb, Ub, bv, Wx);

  // Phase 2: sequential recurrence, ping-pong bf16 hidden state
  for (int t = 0; t < TT; ++t) {
    const __bf16* hp = (t & 1) ? H1 : H0;
    __bf16*       hc = (t & 1) ? H0 : H1;
    rnn_step_kernel<<<DOUT / 128, 256, 0, stream>>>(
        Wx + (size_t)t * BB * DOUT, hp, Uwb,
        out + (size_t)t * BB * DOUT, hc);
  }
}
